// PointNetSeg_89438398972534
// MI455X (gfx1250) — matrix operand order repaired and checked by reading, not compiled, not run on hardware
//
#include <hip/hip_runtime.h>
#include <hip/hip_bf16.h>
#include <math.h>

// ---------------------------------------------------------------------------
// PointNet++ segmentation forward pass for gfx1250 (MI455X).
// All linear layers run through v_wmma_f32_16x16x32_f16 (f16 A/B, f32 acc).
// Weights are staged transposed so both WMMA fragments load from LDS as
// ds_load_b128 (no scalar LDS reads / b16 packing in the inner loop).
// ---------------------------------------------------------------------------

typedef __attribute__((ext_vector_type(16))) _Float16 v16h;
typedef __attribute__((ext_vector_type(8)))  _Float16 h8;
typedef __attribute__((ext_vector_type(8)))  float    v8f;

#define NPTS   16384
#define NGRAPH 8
#define NPER   2048
#define KNBR   16

// ---------------------------------------------------------------------------
// Generic GEMM:  out[M x Np] = A[M x Kp](f16) * W[Kp x Np](f16) + bias[Np]
// WT is W transposed: WT[n][k], row stride Kp.
// Block: 256 threads = 8 waves; block tile 128x16; k-step 32.
// Wave w computes the 16x16 tile at rows [blk*128 + 16w, +16), cols [16*by,+16)
// Requires: M % 128 == 0, Kp % 32 == 0, Np % 16 == 0 (guaranteed by padding).
// ---------------------------------------------------------------------------
template <bool F32OUT>
__global__ __launch_bounds__(256) void gemm_bias(const _Float16* __restrict__ A,
                                                 const _Float16* __restrict__ WT,
                                                 const float* __restrict__ bias,
                                                 void* __restrict__ outv,
                                                 int Kp, int Np) {
  __shared__ _Float16 As[128][40];   // 32 + 8 pad halfs -> 80B stride (16B aligned)
  __shared__ _Float16 BsT[16][40];   // B tile transposed: [n][k], 80B stride

  const int tid  = threadIdx.x;
  const int wave = tid >> 5;
  const int lane = tid & 31;
  const int m    = lane & 15;   // row within A-tile / col within B-tile
  const int kh   = lane >> 4;   // lane-half selector (ISA fragment layout)
  const long row0 = (long)blockIdx.x * 128;
  const int  n0   = blockIdx.y * 16;

  v8f acc = {0.f, 0.f, 0.f, 0.f, 0.f, 0.f, 0.f, 0.f};

  for (int kb = 0; kb < Kp; kb += 32) {
    // Stage A tile: 128x32 halfs as 512 chunks of 8 halfs (16B)
    for (int i = tid; i < 512; i += 256) {
      int r = i >> 2, cc = (i & 3) * 8;
      const _Float16* src = &A[(row0 + r) * (long)Kp + kb + cc];
      *(h8*)&As[r][cc] = *(const h8*)src;
      if (kb + 32 < Kp) __builtin_prefetch(src + 32, 0, 3);  // next k-step tile
    }
    // Stage B tile transposed: 16 cols x 32 k as 64 chunks of 8 halfs.
    // WT rows are contiguous in k, so global loads stay coalesced.
    if (tid < 64) {
      int n = tid >> 2, cc = (tid & 3) * 8;
      *(h8*)&BsT[n][cc] = *(const h8*)&WT[(long)(n0 + n) * Kp + kb + cc];
    }
    __syncthreads();

    // A fragment: lane (m, kh) holds row m, K = kh*8+j and 16+kh*8+j
    // -> two contiguous 16B LDS loads (ds_load_b128)
    const h8 a0 = *(const h8*)&As[wave * 16 + m][kh * 8];
    const h8 a1 = *(const h8*)&As[wave * 16 + m][16 + kh * 8];
    v16h af = __builtin_shufflevector(a0, a1, 0, 1, 2, 3, 4, 5, 6, 7,
                                      8, 9, 10, 11, 12, 13, 14, 15);
    // B fragment: lane (m, kh) holds col m, K = kh*8+j and 16+kh*8+j
    // -> two contiguous 16B LDS loads from the transposed tile
    const h8 b0 = *(const h8*)&BsT[m][kh * 8];
    const h8 b1 = *(const h8*)&BsT[m][16 + kh * 8];
    v16h bf = __builtin_shufflevector(b0, b1, 0, 1, 2, 3, 4, 5, 6, 7,
                                      8, 9, 10, 11, 12, 13, 14, 15);

    acc = __builtin_amdgcn_wmma_f32_16x16x32_f16(false, af, false, bf,
                                                 (short)0, acc, false, false);
    __syncthreads();
  }

  // C layout: VGPR r -> local row (kh*8 + r), col = m
  const long orow = row0 + wave * 16 + kh * 8;
  const float bv  = bias[n0 + m];
  if (F32OUT) {
    float* out = (float*)outv;
#pragma unroll
    for (int r = 0; r < 8; ++r)
      out[(orow + r) * (long)Np + n0 + m] = acc[r] + bv;
  } else {
    _Float16* out = (_Float16*)outv;
#pragma unroll
    for (int r = 0; r < 8; ++r)
      out[(orow + r) * (long)Np + n0 + m] = (_Float16)(acc[r] + bv);
  }
}

// ---------------------------------------------------------------------------
// Column statistics (population mean/var) over M rows of f16 h[M x Np].
// One block per (valid) column.
// ---------------------------------------------------------------------------
__global__ void colstats(const _Float16* __restrict__ h, float* __restrict__ mean,
                         float* __restrict__ var, long M, int Np) {
  const int c = blockIdx.x;
  const int t = threadIdx.x;
  float s = 0.f, s2 = 0.f;
  for (long r = t; r < M; r += blockDim.x) {
    float v = (float)h[r * Np + c];
    s += v; s2 += v * v;
  }
  __shared__ float sh[256], sh2[256];
  sh[t] = s; sh2[t] = s2;
  __syncthreads();
  for (int o = 128; o > 0; o >>= 1) {
    if (t < o) { sh[t] += sh[t + o]; sh2[t] += sh2[t + o]; }
    __syncthreads();
  }
  if (t == 0) {
    float mu = sh[0] / (float)M;
    mean[c] = mu;
    var[c]  = sh2[0] / (float)M - mu * mu;
  }
}

// BN + ReLU in place on f16 buffer; zero the padded columns.
__global__ void bn_relu(_Float16* __restrict__ h, const float* __restrict__ mean,
                        const float* __restrict__ var, const float* __restrict__ gamma,
                        const float* __restrict__ beta, long M, int Np, int Nc) {
  long i = (long)blockIdx.x * blockDim.x + threadIdx.x;
  if (i >= M * (long)Np) return;
  int c = (int)(i % Np);
  if (c >= Nc) { h[i] = (_Float16)0.f; return; }
  float v = (float)h[i];
  v = (v - mean[c]) * rsqrtf(var[c] + 1e-5f) * gamma[c] + beta[c];
  h[i] = (_Float16)fmaxf(v, 0.f);
}

// Per-node max over K neighbors: out[n][c] = max_k h[n*K+k][c]
__global__ void maxagg(const _Float16* __restrict__ h, float* __restrict__ out,
                       int N, int K, int Np, int Nc) {
  long i = (long)blockIdx.x * blockDim.x + threadIdx.x;
  if (i >= (long)N * Nc) return;
  int n = (int)(i / Nc), c = (int)(i % Nc);
  float m = -INFINITY;
  for (int k = 0; k < K; ++k)
    m = fmaxf(m, (float)h[((long)n * K + k) * Np + c]);
  out[(long)n * Nc + c] = m;
}

// Per-graph segment max (batch is contiguous: rows b*NPER .. +NPER)
__global__ void segmax(const _Float16* __restrict__ h, float* __restrict__ out,
                       int nper, int nb, int Np, int Nc) {
  long i = (long)blockIdx.x * blockDim.x + threadIdx.x;
  if (i >= (long)nb * Nc) return;
  int b = (int)(i / Nc), c = (int)(i % Nc);
  float m = -INFINITY;
  for (int r = 0; r < nper; ++r)
    m = fmaxf(m, (float)h[((long)b * nper + r) * Np + c]);
  out[i] = m;
}

// Build PointNetConv messages: out[n*K+k][:] = cat(feat[j], pos[j]-pos[n]), pad 0
__global__ void gather_msg(const float* __restrict__ feat, int C,
                           const float* __restrict__ pos, const int* __restrict__ nbr,
                           int K, _Float16* __restrict__ out, long NK, int Kp) {
  long i = (long)blockIdx.x * blockDim.x + threadIdx.x;
  if (i >= NK * (long)Kp) return;
  long r = i / Kp; int c = (int)(i % Kp);
  long n = r / K;
  int j = nbr[r];
  float v;
  if (c < C)            v = feat[(long)j * C + c];
  else if (c == C)      v = pos[2 * j]     - pos[2 * n];
  else if (c == C + 1)  v = pos[2 * j + 1] - pos[2 * n + 1];
  else                  v = 0.f;
  out[i] = (_Float16)v;
}

// cat(x2[n], pos[n]) padded -> f16
__global__ void gather_cat_pos(const float* __restrict__ x2, int C,
                               const float* __restrict__ pos,
                               _Float16* __restrict__ out, int N, int Kp) {
  long i = (long)blockIdx.x * blockDim.x + threadIdx.x;
  if (i >= (long)N * Kp) return;
  int n = (int)(i / Kp), c = (int)(i % Kp);
  float v = (c < C) ? x2[(long)n * C + c]
          : (c < C + 2 ? pos[2 * n + (c - C)] : 0.f);
  out[i] = (_Float16)v;
}

// FP3 input: cat(xg[batch[n]], x2[n])  (k=1 interp onto centroid == broadcast)
__global__ void gather_fp3(const float* __restrict__ xg, int Cg,
                           const float* __restrict__ x2, int C2,
                           const int* __restrict__ batch,
                           _Float16* __restrict__ out, int N, int Kp) {
  long i = (long)blockIdx.x * blockDim.x + threadIdx.x;
  if (i >= (long)N * Kp) return;
  int n = (int)(i / Kp), c = (int)(i % Kp);
  float v;
  if (c < Cg)           v = xg[(long)batch[n] * Cg + c];
  else if (c < Cg + C2) v = x2[(long)n * C2 + (c - Cg)];
  else                  v = 0.f;
  out[i] = (_Float16)v;
}

// knn-3 inverse-distance weights: w4[n] = {w0,w1,w2,sum}
__global__ void interp_w(const float* __restrict__ pos, const int* __restrict__ knn,
                         float* __restrict__ w4, int N) {
  int n = blockIdx.x * blockDim.x + threadIdx.x;
  if (n >= N) return;
  float px = pos[2 * n], py = pos[2 * n + 1], ws = 0.f;
  for (int k = 0; k < 3; ++k) {
    int j = knn[n * 3 + k];
    float dx = px - pos[2 * j], dy = py - pos[2 * j + 1];
    float w = 1.f / fmaxf(dx * dx + dy * dy, 1e-16f);
    w4[n * 4 + k] = w; ws += w;
  }
  w4[n * 4 + 3] = ws;
}

// FP input: cat(knn_interp(src), skip) padded -> f16
__global__ void gather_interp(const float* __restrict__ src, int Cs,
                              const float* __restrict__ skip, int Ck,
                              const int* __restrict__ knn, const float* __restrict__ w4,
                              _Float16* __restrict__ out, int N, int Kp) {
  long i = (long)blockIdx.x * blockDim.x + threadIdx.x;
  if (i >= (long)N * Kp) return;
  int n = (int)(i / Kp), c = (int)(i % Kp);
  float v;
  if (c < Cs) {
    int i0 = knn[n * 3], i1 = knn[n * 3 + 1], i2 = knn[n * 3 + 2];
    v = (w4[n * 4]     * src[(long)i0 * Cs + c] +
         w4[n * 4 + 1] * src[(long)i1 * Cs + c] +
         w4[n * 4 + 2] * src[(long)i2 * Cs + c]) / w4[n * 4 + 3];
  } else if (c < Cs + Ck) {
    v = skip[(long)n * Ck + (c - Cs)];
  } else v = 0.f;
  out[i] = (_Float16)v;
}

// Weight conversion f32[Kd x Nc] -> TRANSPOSED f16[Np x Kp], zero-padded.
__global__ void cvt_wT(const float* __restrict__ W, _Float16* __restrict__ dst,
                       int Kd, int Nc, int Kp, int Np) {
  long i = (long)blockIdx.x * blockDim.x + threadIdx.x;
  if (i >= (long)Np * Kp) return;
  int n = (int)(i / Kp), k = (int)(i % Kp);
  dst[i] = (k < Kd && n < Nc) ? (_Float16)W[(long)k * Nc + n] : (_Float16)0.f;
}
__global__ void cvt_b(const float* __restrict__ b, float* __restrict__ dst,
                      int Nc, int Np) {
  int i = blockIdx.x * blockDim.x + threadIdx.x;
  if (i >= Np) return;
  dst[i] = (i < Nc) ? b[i] : 0.f;
}

// log_softmax over first C of h[N x Np] -> out[N x C]
__global__ void logsoftmax_rows(const float* __restrict__ h, float* __restrict__ out,
                                int N, int Np, int C) {
  int n = blockIdx.x * blockDim.x + threadIdx.x;
  if (n >= N) return;
  float mx = -INFINITY;
  for (int c = 0; c < C; ++c) mx = fmaxf(mx, h[(long)n * Np + c]);
  float s = 0.f;
  for (int c = 0; c < C; ++c) s += expf(h[(long)n * Np + c] - mx);
  float lse = mx + logf(s);
  for (int c = 0; c < C; ++c) out[(long)n * C + c] = h[(long)n * Np + c] - lse;
}

// ---------------------------------------------------------------------------
// Host-side orchestration
// ---------------------------------------------------------------------------
struct LayerDev {
  const float *W, *b, *beta, *gamma;  // raw f32 params (device)
  _Float16* Wd;                       // padded, transposed f16 weights (ws)
  float* bd;                          // padded f32 bias (ws)
  int Kd, Nc, Kp, Np;
};

// Runs an MLP chain. `in` must differ from bufA. Hidden layers: GEMM(f16 out)
// -> colstats -> BN+ReLU in place; ping-pongs bufA/bufB. Last layer: plain
// linear to lastOut (or next ping buffer if lastOut==nullptr). Returns out ptr.
static void* run_chain(hipStream_t s, LayerDev* L, int nL, long M,
                       const _Float16* in, _Float16* bufA, _Float16* bufB,
                       float* mean, float* var, void* lastOut, bool lastF32) {
  const _Float16* cur = in;
  _Float16* bufs[2] = {bufA, bufB};
  int w = 0;
  void* out = nullptr;
  for (int i = 0; i < nL; ++i) {
    bool last = (i == nL - 1);
    out = (last && lastOut) ? lastOut : (void*)bufs[w];
    dim3 grid((unsigned)(M / 128), (unsigned)(L[i].Np / 16));
    if (last && lastF32)
      gemm_bias<true><<<grid, 256, 0, s>>>(cur, L[i].Wd, L[i].bd, out, L[i].Kp, L[i].Np);
    else
      gemm_bias<false><<<grid, 256, 0, s>>>(cur, L[i].Wd, L[i].bd, out, L[i].Kp, L[i].Np);
    if (!last) {
      colstats<<<L[i].Nc, 256, 0, s>>>((const _Float16*)out, mean, var, M, L[i].Np);
      long tot = M * (long)L[i].Np;
      bn_relu<<<(unsigned)((tot + 255) / 256), 256, 0, s>>>(
          (_Float16*)out, mean, var, L[i].gamma, L[i].beta, M, L[i].Np, L[i].Nc);
      cur = bufs[w];
      w ^= 1;
    }
  }
  return out;
}

extern "C" void kernel_launch(void* const* d_in, const int* in_sizes, int n_in,
                              void* d_out, int out_size, void* d_ws, size_t ws_size,
                              hipStream_t stream) {
  (void)in_sizes; (void)n_in; (void)out_size; (void)ws_size;
  const int N = NPTS;
  const long NK = (long)N * KNBR;

  // ---- input pointer walk --------------------------------------------------
  // Top-level dict insertion order: x, pos, params(tree-flattened), batch,
  // nbr1, nbr2, knn3. params flatten: alphabetical MLP keys
  // (fp1, fp2, fp3, head, sa1, sa2, sa3); per layer sorted keys: W, b [,beta, gamma].
  const float* x   = (const float*)d_in[0];
  const float* pos = (const float*)d_in[1];
  int pi = 2;

  LayerDev Lfp1[3], Lfp2[2], Lfp3[2], Lhead[3], Lsa1[3], Lsa2[3], Lsa3[3];
  auto parse = [&](const int* dims, int nlin, LayerDev* L) {
    for (int i = 0; i < nlin; ++i) {
      L[i].Kd = dims[i]; L[i].Nc = dims[i + 1];
      L[i].W = (const float*)d_in[pi++];
      L[i].b = (const float*)d_in[pi++];
      if (i < nlin - 1) {
        L[i].beta  = (const float*)d_in[pi++];
        L[i].gamma = (const float*)d_in[pi++];
      } else { L[i].beta = nullptr; L[i].gamma = nullptr; }
      L[i].Kp = ((L[i].Kd + 31) / 32) * 32;
      L[i].Np = ((L[i].Nc + 15) / 16) * 16;
    }
  };
  const int d_fp1[]  = {132, 128, 128, 128};
  const int d_fp2[]  = {384, 256, 128};
  const int d_fp3[]  = {1280, 256, 256};
  const int d_head[] = {128, 128, 128, 13};
  const int d_sa1[]  = {6, 64, 64, 128};
  const int d_sa2[]  = {130, 128, 128, 256};
  const int d_sa3[]  = {258, 256, 512, 1024};
  parse(d_fp1, 3, Lfp1);  parse(d_fp2, 2, Lfp2);  parse(d_fp3, 2, Lfp3);
  parse(d_head, 3, Lhead); parse(d_sa1, 3, Lsa1); parse(d_sa2, 3, Lsa2);
  parse(d_sa3, 3, Lsa3);

  const int* batch = (const int*)d_in[pi++];
  const int* nbr1  = (const int*)d_in[pi++];
  const int* nbr2  = (const int*)d_in[pi++];
  const int* knn3  = (const int*)d_in[pi++];

  // ---- workspace layout ----------------------------------------------------
  char* ws = (char*)d_ws;
  size_t off = 0;
  auto alloc = [&](size_t bytes) -> void* {
    off = (off + 255) & ~(size_t)255;
    void* p = ws + off;
    off += bytes;
    return p;
  };
  _Float16* PING = (_Float16*)alloc(NK * 160 * sizeof(_Float16));   // 84 MB
  _Float16* PONG = (_Float16*)alloc(NK * 256 * sizeof(_Float16));   // 134 MB
  float* x1    = (float*)alloc((size_t)N * 128 * 4);
  float* x2    = (float*)alloc((size_t)N * 256 * 4);
  float* xg    = (float*)alloc((size_t)NGRAPH * 1024 * 4);
  float* f3    = (float*)alloc((size_t)N * 256 * 4);
  float* f2    = (float*)alloc((size_t)N * 128 * 4);
  float* headh = (float*)alloc((size_t)N * 16 * 4);
  float* w4    = (float*)alloc((size_t)N * 4 * 4);
  float* mean  = (float*)alloc(1024 * 4);
  float* var   = (float*)alloc(1024 * 4);

  // convert all weights (transposed + padded) / biases (padded)
  auto prep = [&](LayerDev* L, int nlin) {
    for (int i = 0; i < nlin; ++i) {
      L[i].Wd = (_Float16*)alloc((size_t)L[i].Kp * L[i].Np * sizeof(_Float16));
      L[i].bd = (float*)alloc((size_t)L[i].Np * 4);
      long tot = (long)L[i].Kp * L[i].Np;
      cvt_wT<<<(unsigned)((tot + 255) / 256), 256, 0, stream>>>(
          L[i].W, L[i].Wd, L[i].Kd, L[i].Nc, L[i].Kp, L[i].Np);
      cvt_b<<<1, L[i].Np, 0, stream>>>(L[i].b, L[i].bd, L[i].Nc, L[i].Np);
    }
  };
  prep(Lfp1, 3); prep(Lfp2, 2); prep(Lfp3, 2); prep(Lhead, 3);
  prep(Lsa1, 3); prep(Lsa2, 3); prep(Lsa3, 3);

  // ---- SA1: messages [NK x 32] -> MLP -> max over K -> x1 [N x 128] --------
  {
    long tot = NK * 32;
    gather_msg<<<(unsigned)((tot + 255) / 256), 256, 0, stream>>>(
        x, 4, pos, nbr1, KNBR, PING, NK, 32);
    void* h = run_chain(stream, Lsa1, 3, NK, PING, PONG, PING, mean, var,
                        nullptr, false);  // -> PONG f16 [NK x 128]
    long t2 = (long)N * 128;
    maxagg<<<(unsigned)((t2 + 255) / 256), 256, 0, stream>>>(
        (const _Float16*)h, x1, N, KNBR, Lsa1[2].Np, 128);
  }

  // ---- SA2: messages [NK x 160] -> MLP -> max over K -> x2 [N x 256] -------
  {
    long tot = NK * 160;
    gather_msg<<<(unsigned)((tot + 255) / 256), 256, 0, stream>>>(
        x1, 128, pos, nbr2, KNBR, PING, NK, 160);
    void* h = run_chain(stream, Lsa2, 3, NK, PING, PONG, PING, mean, var,
                        nullptr, false);  // -> PONG f16 [NK x 256]
    long t2 = (long)N * 256;
    maxagg<<<(unsigned)((t2 + 255) / 256), 256, 0, stream>>>(
        (const _Float16*)h, x2, N, KNBR, Lsa2[2].Np, 256);
  }

  // ---- SA3: cat(x2,pos) [N x 288] -> MLP -> segment max -> xg [8 x 1024] ---
  {
    long tot = (long)N * 288;
    gather_cat_pos<<<(unsigned)((tot + 255) / 256), 256, 0, stream>>>(
        x2, 256, pos, PING, N, 288);
    void* h = run_chain(stream, Lsa3, 3, N, PING, PONG, PING, mean, var,
                        nullptr, false);  // -> PONG f16 [N x 1024]
    long t2 = (long)NGRAPH * 1024;
    segmax<<<(unsigned)((t2 + 255) / 256), 256, 0, stream>>>(
        (const _Float16*)h, xg, NPER, NGRAPH, Lsa3[2].Np, 1024);
  }

  // ---- FP3: cat(xg[batch], x2) [N x 1280] -> MLP -> f3 (f32) ---------------
  {
    long tot = (long)N * 1280;
    gather_fp3<<<(unsigned)((tot + 255) / 256), 256, 0, stream>>>(
        xg, 1024, x2, 256, batch, PING, N, 1280);
    run_chain(stream, Lfp3, 2, N, PING, PONG, PING, mean, var, f3, true);
  }

  // ---- FP2: cat(interp(f3), x1) [N x 384] -> MLP -> f2 (f32) ---------------
  {
    interp_w<<<(unsigned)((N + 255) / 256), 256, 0, stream>>>(pos, knn3, w4, N);
    long tot = (long)N * 384;
    gather_interp<<<(unsigned)((tot + 255) / 256), 256, 0, stream>>>(
        f3, 256, x1, 128, knn3, w4, PING, N, 384);
    run_chain(stream, Lfp2, 2, N, PING, PONG, PING, mean, var, f2, true);
  }

  // ---- FP1: cat(interp(f2), x) [N x 160] -> MLP -> f1 (f16 in PONG) --------
  void* f1h;
  {
    long tot = (long)N * 160;
    gather_interp<<<(unsigned)((tot + 255) / 256), 256, 0, stream>>>(
        f2, 128, x, 4, knn3, w4, PING, N, 160);
    f1h = run_chain(stream, Lfp1, 3, N, PING, PONG, PING, mean, var,
                    nullptr, false);  // -> PONG f16 [N x 128]
  }

  // ---- HEAD: f1 -> MLP -> headh f32 [N x 16] -> log_softmax -> d_out -------
  {
    run_chain(stream, Lhead, 3, N, (const _Float16*)f1h, PING, PONG, mean, var,
              headh, true);
    logsoftmax_rows<<<(unsigned)((N + 255) / 256), 256, 0, stream>>>(
        headh, (float*)d_out, N, 16, 13);
  }
}
